// GPT_31190052503831
// MI455X (gfx1250) — matrix operand order repaired and checked by reading, not compiled, not run on hardware
//
#include <hip/hip_runtime.h>
#include <hip/hip_bf16.h>
#include <stddef.h>

// ---------------- problem constants (from reference) ----------------
#define NTOK 4096      // B*T = 2*2048
#define CDIM 2048      // C
#define NEXP 64        // E
#define HEXP 256       // H (expert hidden)
#define HSH  512       // HS (shared hidden)
#define TOPK 6
#define CAP  1024

#define LDA 40         // A tile row stride (ushorts): 80B, multiple of 16B
#define LDK 40         // Bt tile col stride (ushorts): 80B, multiple of 16B

typedef __attribute__((ext_vector_type(16))) __bf16 v16bf;
typedef __attribute__((ext_vector_type(8)))  float  v8f;

struct frag_bits { uint4 lo, hi; };   // 32B == sizeof(v16bf)

// ---------------- scalar helpers ----------------
__device__ __forceinline__ unsigned short f32_to_bf16(float f) {
    unsigned int u = __builtin_bit_cast(unsigned int, f);
    unsigned int r = u + 0x7FFFu + ((u >> 16) & 1u);
    return (unsigned short)(r >> 16);
}
__device__ __forceinline__ float fast_sigmoid(float g) {
    return __builtin_amdgcn_rcpf(1.0f + __expf(-g));   // v_rcp_f32, no div ladder
}
__device__ __forceinline__ float silu(float g) {
    return g * fast_sigmoid(g);
}

// ---------------- CDNA5 async global->LDS copy (ASYNCcnt path) ------------
// Generic pointers into the LDS aperture carry the LDS offset in addr[31:0]
// (ISA 10.2), which is exactly what the async op's VDST operand wants.
__device__ __forceinline__ void async_g2lds_b64(void* lds_dst, const void* gsrc) {
    unsigned lds_off = (unsigned)(size_t)lds_dst;
    asm volatile("global_load_async_to_lds_b64 %0, %1, off"
                 :: "v"(lds_off), "v"(gsrc) : "memory");
}
__device__ __forceinline__ void wait_async_zero() {
    asm volatile("s_wait_asynccnt 0x0" ::: "memory");
}

// ---------------- WMMA fragment loaders (2x ds_load_b128 each) ------------
// A: 16x32 bf16 fragment.  LDS tile As[32][LDA].  Lane L: row = mBase + L%16;
// elements 0..7 -> K = (L/16)*8 + 0..7 ; elements 8..15 -> K = (L/16)*8 + 16..23.
__device__ __forceinline__ v16bf load_a_frag(const unsigned short* lds,
                                             int mBase, int lane) {
    const unsigned short* p = lds + (mBase + (lane & 15)) * LDA + ((lane >> 4) * 8);
    frag_bits f;
    f.lo = *(const uint4*)(p);        // 16B aligned: LDA*2=80, (lane>>4)*16 in {0,16}
    f.hi = *(const uint4*)(p + 16);
    return __builtin_bit_cast(v16bf, f);
}
// B: 32x16 bf16 fragment from an N-major (transposed) LDS tile Bt[64][LDK]:
// lane L: col = nBase + L%16; elements 0..7 -> K = (L/16)*8 + 0..7 ; elements 8..15 -> K = (L/16)*8 + 16..23.
__device__ __forceinline__ v16bf load_b_frag(const unsigned short* lds,
                                             int nBase, int lane) {
    const unsigned short* p = lds + (nBase + (lane & 15)) * LDK + ((lane >> 4) * 8);
    frag_bits f;
    f.lo = *(const uint4*)(p);
    f.hi = *(const uint4*)(p + 16);
    return __builtin_bit_cast(v16bf, f);
}
#define WMMA_BF16(acc, a, b) \
    __builtin_amdgcn_wmma_f32_16x16x32_bf16(false, (a), false, (b), (short)0, (acc), false, false)

// ---------------- kernel 1: zero per-expert counters ----------------
__global__ void k_init_counts(int* counts) {
    if (threadIdx.x < NEXP) counts[threadIdx.x] = 0;
}

// ---------------- kernel 2: x f32 -> bf16 ----------------
__global__ void k_cvt_x(const float* __restrict__ x, unsigned short* __restrict__ xb) {
    size_t base = ((size_t)blockIdx.x * 256 + threadIdx.x) * 8;
#pragma unroll
    for (int i = 0; i < 8; ++i) xb[base + i] = f32_to_bf16(x[base + i]);
}

// ---------------- kernel 3: gate: sigmoid scores, top-6, slot assign -------
__global__ void k_gate(const float* __restrict__ x, const float* __restrict__ w_gate,
                       int* counts, int* __restrict__ tok, float* __restrict__ wt) {
    __shared__ float xrow[CDIM];
    __shared__ float scores[NEXP];
    const int t   = blockIdx.x;
    const int tid = threadIdx.x;   // blockDim = 64
    for (int i = tid; i < CDIM; i += 64) xrow[i] = x[(size_t)t * CDIM + i];
    __syncthreads();
    float acc = 0.f;
    const float* wg = w_gate + tid;          // column tid, stride NEXP
    for (int k = 0; k < CDIM; ++k) acc += xrow[k] * wg[(size_t)k * NEXP];
    scores[tid] = fast_sigmoid(acc);
    __syncthreads();
    if (tid == 0) {
        int   idxs[TOPK];
        float vals[TOPK];
        float wsum = 0.f;
#pragma unroll
        for (int k = 0; k < TOPK; ++k) {
            float best = -1.f; int bi = 0;
            for (int e = 0; e < NEXP; ++e)
                if (scores[e] > best) { best = scores[e]; bi = e; }
            scores[bi] = -2.f;
            idxs[k] = bi; vals[k] = best; wsum += best;
        }
        float inv = __builtin_amdgcn_rcpf(wsum);   // ROUTE_SCALE = 1.0
#pragma unroll
        for (int k = 0; k < TOPK; ++k) {
            int e   = idxs[k];
            int pos = atomicAdd(&counts[e], 1);
            if (pos < CAP) {
                tok[e * CAP + pos] = t;
                wt [e * CAP + pos] = vals[k] * inv;
            }
        }
    }
}

// ---------------- staging helpers (run by all 256 threads) -----------------
// A tile: 32 rows x 32 k of bf16, async DMA global->LDS (no VGPR round trip).
__device__ __forceinline__ void stage_a_async(unsigned short* As,
                                              const unsigned short* __restrict__ src,
                                              size_t srcStride, int tid) {
    int i = tid * 4;                  // 1024 ushorts / 256 threads, 8B each
    int r = i >> 5, kk = i & 31;
    async_g2lds_b64(&As[r * LDA + kk], &src[(size_t)r * srcStride + kk]);
}
// B tile (transposed store): 32 k x 64 n from f32 global -> Bt[n][k].
// Coalesced f32 reads, packed b32 LDS stores; prefetch next K-step's rows.
__device__ __forceinline__ void stage_bt(unsigned short* Bt,
                                         const float* __restrict__ w,
                                         size_t ldw, int tid, bool pf) {
#pragma unroll
    for (int p = tid; p < 1024; p += 256) {
        int nn = p & 63, kk = (p >> 6) * 2;
        unsigned int lo = f32_to_bf16(w[(size_t)kk * ldw + nn]);
        unsigned int hi = f32_to_bf16(w[(size_t)(kk + 1) * ldw + nn]);
        *(unsigned int*)&Bt[nn * LDK + kk] = lo | (hi << 16);
    }
    if (pf) __builtin_prefetch(w + 32 * ldw + (size_t)tid * 8, 0, 0);  // next tile
}

// ---------------- kernel 4: shared up-proj + SwiGLU -> h_s bf16 ------------
// su = x @ w_shared_up[C, 2*HS]; h_s[:,j] = silu(su[:,HS+j]) * su[:,j]
__global__ void k_shared_up(const unsigned short* __restrict__ xb,
                            const float* __restrict__ w,
                            unsigned short* __restrict__ hs) {
    __shared__ __align__(16) unsigned short As[32 * LDA];
    __shared__ __align__(16) unsigned short By[64 * LDK];
    __shared__ __align__(16) unsigned short Bg[64 * LDK];
    const int tid = threadIdx.x, lane = tid & 31, wv = tid >> 5;
    const int waveM = wv & 1, waveN = wv >> 1;
    const int m0 = blockIdx.x * 32;
    const int j0 = blockIdx.y * 64;          // in [0, HS)
    v8f accY = {}, accG = {};
    for (int k0 = 0; k0 < CDIM; k0 += 32) {
        __syncthreads();
        stage_a_async(As, xb + (size_t)m0 * CDIM + k0, CDIM, tid);
        bool pf = (k0 + 32) < CDIM;
        stage_bt(By, w + (size_t)k0 * (2 * HSH) + j0,       2 * HSH, tid, pf);
        stage_bt(Bg, w + (size_t)k0 * (2 * HSH) + HSH + j0, 2 * HSH, tid, pf);
        wait_async_zero();
        __syncthreads();
        v16bf a  = load_a_frag(As, waveM * 16, lane);
        v16bf by = load_b_frag(By, waveN * 16, lane);
        v16bf bg = load_b_frag(Bg, waveN * 16, lane);
        accY = WMMA_BF16(accY, a, by);
        accG = WMMA_BF16(accG, a, bg);
    }
    int row = m0 + waveM * 16 + ((lane >> 4) * 8);
    int col = j0 + waveN * 16 + (lane & 15);
#pragma unroll
    for (int r = 0; r < 8; ++r)
        hs[(size_t)(row + r) * HSH + col] = f32_to_bf16(silu(accG[r]) * accY[r]);
}

// ---------------- kernel 5: shared down-proj -> writes d_out ---------------
__global__ void k_shared_down(const unsigned short* __restrict__ hs,
                              const float* __restrict__ w,   // [HS, C]
                              float* __restrict__ out) {
    __shared__ __align__(16) unsigned short As[32 * LDA];
    __shared__ __align__(16) unsigned short Bt[64 * LDK];
    const int tid = threadIdx.x, lane = tid & 31, wv = tid >> 5;
    const int waveM = wv & 1, waveN = wv >> 1;
    const int m0 = blockIdx.x * 32;
    const int n0 = blockIdx.y * 64;
    v8f acc = {};
    for (int k0 = 0; k0 < HSH; k0 += 32) {
        __syncthreads();
        stage_a_async(As, hs + (size_t)m0 * HSH + k0, HSH, tid);
        stage_bt(Bt, w + (size_t)k0 * CDIM + n0, CDIM, tid, (k0 + 32) < HSH);
        wait_async_zero();
        __syncthreads();
        v16bf a = load_a_frag(As, waveM * 16, lane);
        v16bf b = load_b_frag(Bt, waveN * 16, lane);
        acc = WMMA_BF16(acc, a, b);
    }
    int row = m0 + waveM * 16 + ((lane >> 4) * 8);
    int col = n0 + waveN * 16 + (lane & 15);
#pragma unroll
    for (int r = 0; r < 8; ++r)
        out[(size_t)(row + r) * CDIM + col] = acc[r];
}

// ---------------- kernel 6: expert up-proj + SwiGLU (gathered rows) --------
// up = X_e @ w_up[e][C, 2H]; h = silu(up[:, :H]) * up[:, H:]
// Invalid rows (>= cnt) are redirected to token 0: each A row only feeds its
// own output row, and those rows are masked out at the store.
__global__ void k_expert_up(const unsigned short* __restrict__ xb,
                            const float* __restrict__ w_up,  // [E, C, 2H]
                            const int* __restrict__ counts,
                            const int* __restrict__ tok,
                            unsigned short* __restrict__ hr) {
    __shared__ __align__(16) unsigned short As[32 * LDA];
    __shared__ __align__(16) unsigned short Bg[64 * LDK];
    __shared__ __align__(16) unsigned short Bv[64 * LDK];
    __shared__ int tokLds[32];
    const int e   = blockIdx.x;
    int cnt = counts[e]; cnt = cnt > CAP ? CAP : cnt;
    const int r0  = blockIdx.y * 32;
    if (r0 >= cnt) return;
    const int j0  = blockIdx.z * 64;         // in [0, H)
    const int tid = threadIdx.x, lane = tid & 31, wv = tid >> 5;
    const int waveM = wv & 1, waveN = wv >> 1;
    if (tid < 32) {
        int rg = r0 + tid;
        tokLds[tid] = (rg < cnt) ? tok[e * CAP + rg] : 0;   // redirect, not zero-fill
    }
    const float* wbase = w_up + (size_t)e * CDIM * (2 * HEXP);
    v8f accG = {}, accV = {};
    for (int k0 = 0; k0 < CDIM; k0 += 32) {
        __syncthreads();
        {   // gathered async A stage: one b64 DMA per thread
            int i = tid * 4;
            int r = i >> 5, kk = i & 31;
            int t = tokLds[r];
            async_g2lds_b64(&As[r * LDA + kk], &xb[(size_t)t * CDIM + k0 + kk]);
        }
        bool pf = (k0 + 32) < CDIM;
        stage_bt(Bg, wbase + (size_t)k0 * (2 * HEXP) + j0,        2 * HEXP, tid, pf);
        stage_bt(Bv, wbase + (size_t)k0 * (2 * HEXP) + HEXP + j0, 2 * HEXP, tid, pf);
        wait_async_zero();
        __syncthreads();
        v16bf a  = load_a_frag(As, waveM * 16, lane);
        v16bf bg = load_b_frag(Bg, waveN * 16, lane);
        v16bf bv = load_b_frag(Bv, waveN * 16, lane);
        accG = WMMA_BF16(accG, a, bg);
        accV = WMMA_BF16(accV, a, bv);
    }
    int rowl = waveM * 16 + ((lane >> 4) * 8);
    int col  = j0 + waveN * 16 + (lane & 15);
#pragma unroll
    for (int r = 0; r < 8; ++r) {
        int row = r0 + rowl + r;
        if (row < cnt)
            hr[((size_t)e * CAP + row) * HEXP + col] = f32_to_bf16(silu(accG[r]) * accV[r]);
    }
}

// ---------------- kernel 7: expert down-proj + weighted atomic combine -----
// Rows >= cnt read stale h_r values (finite bf16), results discarded at store.
__global__ void k_expert_down(const unsigned short* __restrict__ hr,
                              const float* __restrict__ w_down, // [E, H, C]
                              const int* __restrict__ counts,
                              const int* __restrict__ tok,
                              const float* __restrict__ wt,
                              float* __restrict__ out) {
    __shared__ __align__(16) unsigned short As[32 * LDA];
    __shared__ __align__(16) unsigned short Bt[64 * LDK];
    __shared__ int   tokLds[32];
    __shared__ float wtLds[32];
    const int e   = blockIdx.x;
    int cnt = counts[e]; cnt = cnt > CAP ? CAP : cnt;
    const int r0  = blockIdx.y * 32;
    if (r0 >= cnt) return;
    const int n0  = blockIdx.z * 64;
    const int tid = threadIdx.x, lane = tid & 31, wv = tid >> 5;
    const int waveM = wv & 1, waveN = wv >> 1;
    if (tid < 32) {
        int rg = r0 + tid;
        bool v = rg < cnt;
        tokLds[tid] = v ? tok[e * CAP + rg] : -1;
        wtLds[tid]  = v ? wt[e * CAP + rg] : 0.f;
    }
    const float* wbase = w_down + (size_t)e * HEXP * CDIM;
    v8f acc = {};
    for (int k0 = 0; k0 < HEXP; k0 += 32) {
        __syncthreads();
        {   // async A stage from h_r (unconditional; tile stays in-bounds)
            int i = tid * 4;
            int r = i >> 5, kk = i & 31;
            async_g2lds_b64(&As[r * LDA + kk],
                            &hr[((size_t)e * CAP + r0 + r) * HEXP + k0 + kk]);
        }
        stage_bt(Bt, wbase + (size_t)k0 * CDIM + n0, CDIM, tid, (k0 + 32) < HEXP);
        wait_async_zero();
        __syncthreads();
        v16bf a = load_a_frag(As, waveM * 16, lane);
        v16bf b = load_b_frag(Bt, waveN * 16, lane);
        acc = WMMA_BF16(acc, a, b);
    }
    int rowl = waveM * 16 + ((lane >> 4) * 8);
    int col  = n0 + waveN * 16 + (lane & 15);
#pragma unroll
    for (int r = 0; r < 8; ++r) {
        int rl  = rowl + r;
        int row = r0 + rl;
        if (row < cnt) {
            int   t  = tokLds[rl];
            float ww = wtLds[rl];
            atomicAdd(&out[(size_t)t * CDIM + col], ww * acc[r]);
        }
    }
}

// ---------------- workspace layout ----------------
static const size_t OFF_COUNTS = 0;                                   // 64*4 -> pad 1KB
static const size_t OFF_TOK    = 1024;                                // E*CAP*4
static const size_t OFF_WT     = OFF_TOK + (size_t)NEXP * CAP * 4;    // E*CAP*4
static const size_t OFF_XBF    = OFF_WT  + (size_t)NEXP * CAP * 4;    // N*C*2
static const size_t OFF_HS     = OFF_XBF + (size_t)NTOK * CDIM * 2;   // N*HS*2
static const size_t OFF_HR     = OFF_HS  + (size_t)NTOK * HSH  * 2;   // E*CAP*H*2

extern "C" void kernel_launch(void* const* d_in, const int* in_sizes, int n_in,
                              void* d_out, int out_size, void* d_ws, size_t ws_size,
                              hipStream_t stream) {
    (void)in_sizes; (void)n_in; (void)out_size; (void)ws_size;
    const float* x        = (const float*)d_in[0];
    const float* w_gate   = (const float*)d_in[1];
    const float* w_sh_up  = (const float*)d_in[2];
    const float* w_sh_dn  = (const float*)d_in[3];
    const float* w_up     = (const float*)d_in[4];
    const float* w_down   = (const float*)d_in[5];
    float* out = (float*)d_out;
    char*  ws  = (char*)d_ws;

    int*            counts = (int*)(ws + OFF_COUNTS);
    int*            tok    = (int*)(ws + OFF_TOK);
    float*          wt     = (float*)(ws + OFF_WT);
    unsigned short* xbf    = (unsigned short*)(ws + OFF_XBF);
    unsigned short* hs     = (unsigned short*)(ws + OFF_HS);
    unsigned short* hr     = (unsigned short*)(ws + OFF_HR);

    k_init_counts<<<1, 64, 0, stream>>>(counts);
    k_cvt_x<<<(NTOK * CDIM) / (256 * 8), 256, 0, stream>>>(x, xbf);
    k_gate<<<NTOK, 64, 0, stream>>>(x, w_gate, counts, tok, wt);
    k_shared_up<<<dim3(NTOK / 32, HSH / 64), 256, 0, stream>>>(xbf, w_sh_up, hs);
    k_shared_down<<<dim3(NTOK / 32, CDIM / 64), 256, 0, stream>>>(hs, w_sh_dn, out);
    k_expert_up<<<dim3(NEXP, CAP / 32, HEXP / 64), 256, 0, stream>>>(xbf, w_up, counts, tok, hr);
    k_expert_down<<<dim3(NEXP, CAP / 32, CDIM / 64), 256, 0, stream>>>(hr, w_down, counts, tok, wt, out);
}
